// CGNN_New_43499428774090
// MI455X (gfx1250) — matrix operand order repaired and checked by reading, not compiled, not run on hardware
//
#include <hip/hip_runtime.h>

typedef __attribute__((ext_vector_type(16))) __bf16 v16bf;
typedef __attribute__((ext_vector_type(8)))  __bf16 v8bf;
typedef __attribute__((ext_vector_type(8)))  float  v8f;
typedef __attribute__((ext_vector_type(4)))  unsigned int v4u;
typedef __attribute__((ext_vector_type(8)))  int v8i;
typedef __attribute__((ext_vector_type(4)))  int v4i;

#define NEG_SLOPE 0.01f

// ---------------- degree / normalization ----------------

__global__ void k_zero_f32(float* p, long long n) {
    long long i = (long long)blockIdx.x * blockDim.x + threadIdx.x;
    if (i < n) p[i] = 0.0f;
}

__global__ void k_deg_count(const int* __restrict__ dst, float* __restrict__ deg, int E) {
    int i = blockIdx.x * blockDim.x + threadIdx.x;
    if (i < E) atomicAdd(&deg[dst[i]], 1.0f);
}

// in-place: deg -> dis = rsqrt(deg + 1)   (+1 = self loop; always > 0)
__global__ void k_deg_to_dis(float* __restrict__ d, int n) {
    int i = blockIdx.x * blockDim.x + threadIdx.x;
    if (i < n) d[i] = rsqrtf(d[i] + 1.0f);
}

// ---------------- bf16 casts ----------------

__global__ void k_cast_bf16(const float* __restrict__ in, __bf16* __restrict__ out, long long n) {
    long long i = (long long)blockIdx.x * blockDim.x + threadIdx.x;
    if (i < n) out[i] = (__bf16)in[i];
}

// W[K][Nout] (row major) -> WT[Nout][K] bf16
__global__ void k_cast_wT(const float* __restrict__ W, __bf16* __restrict__ WT, int K, int Nout) {
    int gid = blockIdx.x * blockDim.x + threadIdx.x;
    if (gid < K * Nout) {
        int k = gid / Nout;
        int n = gid % Nout;
        WT[(long long)n * K + k] = (__bf16)W[gid];
    }
}

// ---------------- WMMA GEMM:  C[M][Nout] = A[M][K](bf16) * WT[Nout][K](bf16) ----------------
// Block = 128 rows (8 waves x 16-row strips). The block's A panel (128 x K bf16,
// <= 64KB) is DMA'd into LDS ONCE by the Tensor Data Mover (wave 0 builds a D#
// descriptor per CDNA5 ISA ch.8, issues tensor_load_to_lds, waits TENSORcnt).
// tensor_dim1 = rows remaining, tile_dim1 = 128 -> TDM zero-fills OOB rows, so
// fragment loads need no clamping. A-fragments then come from LDS (ds_load_b128),
// B-fragments are contiguous 32B global loads from the pre-transposed weights.
// Fragment layouts used here (contiguous k, not the ISA table order) (hh = lane>>4, rl = lane&15):
//   A (16x32): lane holds row rl; K = k0+16*hh+{0..15}
//   B (32x16): lane holds col rl; K = k0+16*hh+{0..15}
__global__ void k_gemm_bf16_wmma(const __bf16* __restrict__ A, const __bf16* __restrict__ BT,
                                 float* __restrict__ C, int M, int K, int Nout) {
    __shared__ __bf16 sA[128 * 256];

    const int wave = threadIdx.x >> 5;
    const int lane = threadIdx.x & 31;
    const int rl = lane & 15;          // row-in-tile (A) / col-in-tile (B,C)
    const int hh = lane >> 4;          // half select

    const int blk0 = blockIdx.x * 128;

    if (threadIdx.x < 32) {            // wave 0 issues the TDM load (EXEC ignored by TDM)
        unsigned lds_base = (unsigned)(unsigned long long)(uintptr_t)sA; // LDS byte offset
        unsigned long long ga =
            (unsigned long long)(uintptr_t)(A + (long long)blk0 * K);    // tile start (bytes)
        int rowsAvail = M - blk0;      // tensor_dim1: rows past this -> zero-fill

        v4u g0;
        g0[0] = 1u;                                              // count=1 (valid user D#)
        g0[1] = lds_base;                                        // lds_addr
        g0[2] = (unsigned)ga;                                    // global_addr[31:0]
        g0[3] = (unsigned)((ga >> 32) & 0x01FFFFFFu) | (2u << 30); // addr[56:32] | type=2

        v8i g1;
        g1[0] = 1 << 16;                                         // data_size=1 (2 bytes)
        g1[1] = (K & 0xFFFF) << 16;                              // tensor_dim0[15:0]
        g1[2] = ((K >> 16) & 0xFFFF) | ((rowsAvail & 0xFFFF) << 16); // dim0 hi | dim1 lo
        g1[3] = ((rowsAvail >> 16) & 0xFFFF) | ((K & 0xFFFF) << 16); // dim1 hi | tile_dim0=K
        g1[4] = 128;                                             // tile_dim1=128 (tile_dim2=0)
        g1[5] = K;                                               // tensor_dim0_stride lo32
        g1[6] = 0;                                               // stride hi / dim1_stride lo
        g1[7] = 0;

        v4i g2 = {0, 0, 0, 0};                                   // dims 2-4 unused
        v4i g3 = {0, 0, 0, 0};
#if __clang_major__ >= 23
        v8i g4 = {0, 0, 0, 0, 0, 0, 0, 0};
        __builtin_amdgcn_tensor_load_to_lds(g0, g1, g2, g3, g4, 0);
#else
        __builtin_amdgcn_tensor_load_to_lds(g0, g1, g2, g3, 0);
#endif
        __builtin_amdgcn_s_wait_tensorcnt(0);
    }
    __syncthreads();                   // publish LDS panel to all 8 waves

    const __bf16* aldsrow = sA + (wave * 16 + rl) * K + hh * 16;

    const int ntiles = Nout >> 4;
    for (int nt = 0; nt < ntiles; ++nt) {
        const int col = nt * 16 + rl;
        const __bf16* bcol = BT + (long long)col * K + hh * 16;

        v8f c = {};
        for (int k0 = 0; k0 < K; k0 += 32) {
            v8bf alo = *(const v8bf*)(aldsrow + k0);       // ds_load_b128
            v8bf ahi = *(const v8bf*)(aldsrow + k0 + 8);   // ds_load_b128
            v16bf a;
#pragma unroll
            for (int i = 0; i < 8; ++i) { a[i] = alo[i]; a[i + 8] = ahi[i]; }
            v16bf b = *(const v16bf*)(bcol + k0);          // 2x global_load_b128
            c = __builtin_amdgcn_wmma_f32_16x16x32_bf16(
                    false, a, false, b, (short)0, c, false, false);
        }
#pragma unroll
        for (int r = 0; r < 8; ++r) {  // VGPR r: lanes0-15 -> M=r, lanes16-31 -> M=8+r
            int orow = blk0 + wave * 16 + r + hh * 8;
            if (orow < M) C[(long long)orow * Nout + nt * 16 + rl] = c[r];
        }
    }
}

// ---------------- aggregation ----------------

// agg[i][j] = dis[i]^2 * h[i][j] + bias[j]   (self-loop term + bias)
__global__ void k_agg_init(const float* __restrict__ h, const float* __restrict__ dis,
                           const float* __restrict__ bias, float* __restrict__ agg,
                           long long total, int nshift, int nmask) {
    long long gid = (long long)blockIdx.x * blockDim.x + threadIdx.x;
    if (gid >= total) return;
    int i = (int)(gid >> nshift);
    int j = (int)(gid & nmask);
    float di = dis[i];
    agg[gid] = di * di * h[gid] + bias[j];
}

// edge-parallel: agg[dst] += dis[src]*dis[dst] * h[src]  (float4 chunks, L2 atomics)
__global__ void k_edge_scatter(const int* __restrict__ src, const int* __restrict__ dst,
                               const float* __restrict__ dis, const float* __restrict__ h,
                               float* __restrict__ agg, long long total, int c4shift, int Nout) {
    long long gid = (long long)blockIdx.x * blockDim.x + threadIdx.x;
    if (gid >= total) return;
    long long e = gid >> c4shift;
    int ch = (int)(gid & ((1 << c4shift) - 1)) * 4;
    int s = src[e];
    int d = dst[e];
    float w = dis[s] * dis[d];
    float4 v = *(const float4*)(h + (long long)s * Nout + ch);
    float* ap = agg + (long long)d * Nout + ch;
    atomicAdd(ap + 0, w * v.x);
    atomicAdd(ap + 1, w * v.y);
    atomicAdd(ap + 2, w * v.z);
    atomicAdd(ap + 3, w * v.w);
}

// leaky-relu + cast to bf16 (input of next layer's GEMM)
__global__ void k_lrelu_cast(const float* __restrict__ in, __bf16* __restrict__ out, long long n) {
    long long i = (long long)blockIdx.x * blockDim.x + threadIdx.x;
    if (i < n) {
        float v = in[i];
        out[i] = (__bf16)(v > 0.0f ? v : NEG_SLOPE * v);
    }
}

// final head: out[i] = leaky(agg3[i]) . Wfc + bfc   (dim 32 -> 1)
__global__ void k_fc(const float* __restrict__ agg, const float* __restrict__ Wfc,
                     const float* __restrict__ bfc, float* __restrict__ out, int n) {
    int i = blockIdx.x * blockDim.x + threadIdx.x;
    if (i >= n) return;
    const float* r = agg + (long long)i * 32;
    float acc = bfc[0];
#pragma unroll
    for (int j = 0; j < 32; ++j) {
        float v = r[j];
        v = v > 0.0f ? v : NEG_SLOPE * v;
        acc += v * Wfc[j];
    }
    out[i] = acc;
}

// ---------------- host ----------------

static inline unsigned blks(long long n, int t) { return (unsigned)((n + t - 1) / t); }

extern "C" void kernel_launch(void* const* d_in, const int* in_sizes, int n_in,
                              void* d_out, int out_size, void* d_ws, size_t ws_size,
                              hipStream_t stream) {
    const float* x   = (const float*)d_in[0];
    const int*   ei  = (const int*)d_in[1];
    const float* W1  = (const float*)d_in[2]; const float* b1  = (const float*)d_in[3];
    const float* W2  = (const float*)d_in[4]; const float* b2  = (const float*)d_in[5];
    const float* W3  = (const float*)d_in[6]; const float* b3  = (const float*)d_in[7];
    const float* Wfc = (const float*)d_in[8]; const float* bfc = (const float*)d_in[9];

    const int N = in_sizes[0] / 128;
    const int E = in_sizes[1] / 2;
    const int* src = ei;
    const int* dst = ei + E;

    // carve workspace (256B aligned)
    char* wsp = (char*)d_ws;
    auto carve = [&](size_t bytes) -> void* {
        void* p = (void*)wsp;
        wsp += (bytes + 255) & ~(size_t)255;
        return p;
    };
    float*  dis = (float*) carve((size_t)N * 4);            // degree, then rsqrt
    __bf16* xb  = (__bf16*)carve((size_t)N * 256 * 2);      // bf16 activations (GEMM A)
    __bf16* wbT = (__bf16*)carve((size_t)256 * 256 * 2);    // bf16 transposed weights
    float*  hw  = (float*) carve((size_t)N * 256 * 4);      // GEMM output h = A*W
    float*  agg = (float*) carve((size_t)N * 256 * 4);      // aggregated output
    float*  out = (float*)d_out;

    const int T = 256;

    // degrees -> dis
    k_zero_f32<<<blks(N, T), T, 0, stream>>>(dis, N);
    k_deg_count<<<blks(E, T), T, 0, stream>>>(dst, dis, E);
    k_deg_to_dis<<<blks(N, T), T, 0, stream>>>(dis, N);

    // layer driver
    auto layer = [&](const __bf16* Ain, const float* W, const float* bias,
                     int K, int Nout, int nshift, int c4shift) {
        k_cast_wT<<<blks((long long)K * Nout, T), T, 0, stream>>>(W, wbT, K, Nout);
        k_gemm_bf16_wmma<<<blks(N, 128), T, 0, stream>>>(Ain, wbT, hw, N, K, Nout);
        long long tot = (long long)N * Nout;
        k_agg_init<<<blks(tot, T), T, 0, stream>>>(hw, dis, bias, agg, tot, nshift, Nout - 1);
        long long etot = (long long)E << c4shift;
        k_edge_scatter<<<blks(etot, T), T, 0, stream>>>(src, dst, dis, hw, agg, etot, c4shift, Nout);
    };

    // layer 1: 128 -> 256
    k_cast_bf16<<<blks((long long)N * 128, T), T, 0, stream>>>(x, xb, (long long)N * 128);
    layer(xb, W1, b1, 128, 256, 8, 6);
    k_lrelu_cast<<<blks((long long)N * 256, T), T, 0, stream>>>(agg, xb, (long long)N * 256);

    // layer 2: 256 -> 128
    layer(xb, W2, b2, 256, 128, 7, 5);
    k_lrelu_cast<<<blks((long long)N * 128, T), T, 0, stream>>>(agg, xb, (long long)N * 128);

    // layer 3: 128 -> 32
    layer(xb, W3, b3, 128, 32, 5, 3);

    // head (fuses last leaky-relu): 32 -> 1
    k_fc<<<blks(N, T), T, 0, stream>>>(agg, Wfc, bfc, out, N);
}
